// MLP_32289564131760
// MI455X (gfx1250) — matrix operand order repaired and checked by reading, not compiled, not run on hardware
//
#include <hip/hip_runtime.h>
#include <hip/hip_bf16.h>

typedef __bf16 bf16_t;
typedef __attribute__((ext_vector_type(16))) __bf16 v16bf;
typedef __attribute__((ext_vector_type(8)))  __bf16 v8bf;
typedef __attribute__((ext_vector_type(8)))  float  v8f;

#define B_ROWS     16384
#define D_EMB      64
#define RANK       16
#define SCALE_LORA 2.0f
#define GDIM       256

// GEMM tiling: block = 256x128 out tile, K-step 32 (bf16 WMMA K), 8 wave32 waves
// arranged 4(M) x 2(N); each wave computes a 64x64 sub-tile = 16 WMMA tiles.
#define BM 256
#define BN 128
#define BK 32
#define LDS_STRIDE 40   // BK + 8 pad elems => 80B row stride, keeps 16B alignment

// ---------------------------------------------------------------------------
// Embedding gather: v0[b, f*64+o] = emb_f[idx_f[b]][o]  -> bf16
// ---------------------------------------------------------------------------
__global__ void embed_gather_kernel(const int* __restrict__ i0, const int* __restrict__ i1,
                                    const int* __restrict__ i2, const int* __restrict__ i3,
                                    const float* __restrict__ e0, const float* __restrict__ e1,
                                    const float* __restrict__ e2, const float* __restrict__ e3,
                                    bf16_t* __restrict__ v0) {
  int id = blockIdx.x * blockDim.x + threadIdx.x;   // B*256 threads
  int b = id >> 8, c = id & 255;
  int f = c >> 6, off = c & 63;
  const float* e; int idx;
  if      (f == 0) { e = e0; idx = i0[b]; }
  else if (f == 1) { e = e1; idx = i1[b]; }
  else if (f == 2) { e = e2; idx = i2[b]; }
  else             { e = e3; idx = i3[b]; }
  v0[id] = (bf16_t)e[(long)idx * D_EMB + off];
}

// ---------------------------------------------------------------------------
// Convert + transpose weights: W[K,N] f32 -> Wt[N,K] bf16 (contiguous K per col)
// ---------------------------------------------------------------------------
__global__ void convt_kernel(const float* __restrict__ W, bf16_t* __restrict__ Wt,
                             int K, int N) {
  long id = (long)blockIdx.x * blockDim.x + threadIdx.x;
  if (id >= (long)K * N) return;
  int k = (int)(id / N), n = (int)(id % N);
  Wt[(long)n * K + k] = (bf16_t)W[id];
}

// ---------------------------------------------------------------------------
// bf16 WMMA GEMM: Out[M,N] = epilogue(A[M,K] @ Wt[N,K]^T + bias)
// MODE 0: Out = relu(acc + bias)                              (gate pre-act)
// MODE 1: Out = relu(acc+bias)*g0 + SCALE*((tu@lbu)*g1 + (tf@lbf)*g2)  (fc)
// Tiles are staged memory->LDS with GLOBAL_LOAD_ASYNC_TO_LDS_B128 (ASYNCcnt).
// ---------------------------------------------------------------------------
template <int MODE>
__global__ __launch_bounds__(256)
void gemm_bf16_kernel(const bf16_t* __restrict__ A, const bf16_t* __restrict__ Wt,
                      const float* __restrict__ bias, bf16_t* __restrict__ Out,
                      int K, int N,
                      const float* __restrict__ gate,
                      const float* __restrict__ tu, const float* __restrict__ tf,
                      const bf16_t* __restrict__ lbu_t, const bf16_t* __restrict__ lbf_t)
{
  __shared__ __attribute__((aligned(16))) bf16_t As[2][BM * LDS_STRIDE];
  __shared__ __attribute__((aligned(16))) bf16_t Bs[2][BN * LDS_STRIDE];

  const int tid  = threadIdx.x;
  const int wave = tid >> 5;
  const int lane = tid & 31;
  const int half = lane >> 4;     // K-half (A/B frags) / M+8 select (C frag)
  const int lm   = lane & 15;     // M within A tile, N within B/C tile

  const int wm = (wave & 3) * 64; // wave M offset in block tile (4 waves in M)
  const int wn = (wave >> 2) * 64;// wave N offset (2 waves in N)

  const int rowBase = blockIdx.y * BM;
  const int colBase = blockIdx.x * BN;

  const bf16_t* aPtr = A  + (long)rowBase * K;
  const bf16_t* bPtr = Wt + (long)colBase * K;

  // Async copy of one K-slab (A: 256x32, B: 128x32) into LDS buffer `buf`.
  // Each chunk is 8 bf16 = 16B; lane supplies its own LDS dest address
  // (low 32 bits of the generic pointer == LDS byte address per aperture rules).
  auto asyncLoadTile = [&](int buf, int k0) {
#pragma unroll
    for (int i = 0; i < 4; ++i) {        // A tile: 1024 chunks / 256 threads
      int c = tid + i * 256;
      int r = c >> 2, c8 = (c & 3) * 8;
      unsigned lds = (unsigned)(unsigned long)&As[buf][r * LDS_STRIDE + c8];
      const bf16_t* g = aPtr + (long)r * K + k0 + c8;
      asm volatile("global_load_async_to_lds_b128 %0, %1, off"
                   :: "v"(lds), "v"(g) : "memory");
    }
#pragma unroll
    for (int i = 0; i < 2; ++i) {        // B tile: 512 chunks / 256 threads
      int c = tid + i * 256;
      int r = c >> 2, c8 = (c & 3) * 8;
      unsigned lds = (unsigned)(unsigned long)&Bs[buf][r * LDS_STRIDE + c8];
      const bf16_t* g = bPtr + (long)r * K + k0 + c8;
      asm volatile("global_load_async_to_lds_b128 %0, %1, off"
                   :: "v"(lds), "v"(g) : "memory");
    }
  };
  auto waitAsync = [&]() {
    asm volatile("s_wait_asynccnt 0x0" ::: "memory");
  };

  v8f acc[4][4];
#pragma unroll
  for (int i = 0; i < 4; ++i)
#pragma unroll
    for (int j = 0; j < 4; ++j)
#pragma unroll
      for (int e = 0; e < 8; ++e) acc[i][j][e] = 0.0f;

  const int nk = K / BK;
  asyncLoadTile(0, 0);
  waitAsync();
  __syncthreads();

  for (int kt = 0; kt < nk; ++kt) {
    const int cur = kt & 1;
    if (kt + 1 < nk) {
      asyncLoadTile(cur ^ 1, (kt + 1) * BK);     // in flight during compute
      if (kt + 2 < nk) {  // L2 prefetch of tile kt+2 (global_prefetch_b8)
        __builtin_prefetch(aPtr + (long)(tid >> 1) * K + (kt + 2) * BK, 0, 1);
        __builtin_prefetch(bPtr + (long)(tid >> 2) * K + (kt + 2) * BK, 0, 1);
      }
    }

    // A fragment: lane holds row M=lm; K = half*8..+7 then 16+half*8..+7 (ISA layout)
    v16bf afrag[4], bfrag[4];
#pragma unroll
    for (int i = 0; i < 4; ++i) {
      const bf16_t* p = &As[cur][(wm + i * 16 + lm) * LDS_STRIDE];
      ((v8bf*)&afrag[i])[0] = *(const v8bf*)(p + half * 8);
      ((v8bf*)&afrag[i])[1] = *(const v8bf*)(p + 16 + half * 8);
    }
    // B fragment: lane holds col N=lm; K = half*8..+7 then 16+half*8..+7 (same as A)
#pragma unroll
    for (int j = 0; j < 4; ++j) {
      const bf16_t* p = &Bs[cur][(wn + j * 16 + lm) * LDS_STRIDE + half * 8];
      ((v8bf*)&bfrag[j])[0] = *(const v8bf*)(p);
      ((v8bf*)&bfrag[j])[1] = *(const v8bf*)(p + 16);
    }

#pragma unroll
    for (int i = 0; i < 4; ++i)
#pragma unroll
      for (int j = 0; j < 4; ++j)
        acc[i][j] = __builtin_amdgcn_wmma_f32_16x16x32_bf16(
            false, afrag[i], false, bfrag[j], (short)0, acc[i][j], false, false);

    if (kt + 1 < nk) {
      waitAsync();          // next slab fully in LDS
      __syncthreads();
    }
  }

  // Epilogue. C layout: VGPR r -> M = r + 8*half, N = lm.
#pragma unroll
  for (int i = 0; i < 4; ++i) {
#pragma unroll
    for (int j = 0; j < 4; ++j) {
      const int col = colBase + wn + j * 16 + lm;
      const float bcol = bias[col];
#pragma unroll
      for (int r = 0; r < 8; ++r) {
        const int row = rowBase + wm + i * 16 + r + half * 8;
        float v = acc[i][j][r] + bcol;
        if (MODE == 0) {
          Out[(long)row * N + col] = (bf16_t)fmaxf(v, 0.0f);
        } else {
          const float* g      = gate + row * 4;
          const float* turow  = tu + row * RANK;
          const float* tfrow  = tf + row * RANK;
          const bf16_t* bu    = lbu_t + (long)col * RANK;
          const bf16_t* bfp   = lbf_t + (long)col * RANK;
          float lu = 0.0f, lf = 0.0f;
#pragma unroll
          for (int q = 0; q < RANK; ++q) {
            lu += turow[q] * (float)bu[q];
            lf += tfrow[q] * (float)bfp[q];
          }
          float val = fmaxf(v, 0.0f) * g[0] + SCALE_LORA * (lu * g[1] + lf * g[2]);
          Out[(long)row * N + col] = (bf16_t)val;
        }
      }
    }
  }
}

// ---------------------------------------------------------------------------
// Gate tail: g = softmax(U[B,256] @ g2w[256,3] + g2b). One wave32 per row.
// ---------------------------------------------------------------------------
__global__ void gate_kernel(const bf16_t* __restrict__ U, const float* __restrict__ g2w,
                            const float* __restrict__ g2b, float* __restrict__ G) {
  int row  = (blockIdx.x * blockDim.x + threadIdx.x) >> 5;
  int lane = threadIdx.x & 31;
  float s0 = 0.f, s1 = 0.f, s2 = 0.f;
  for (int k = lane; k < GDIM; k += 32) {
    float u = (float)U[(long)row * GDIM + k];
    s0 += u * g2w[k * 3 + 0];
    s1 += u * g2w[k * 3 + 1];
    s2 += u * g2w[k * 3 + 2];
  }
#pragma unroll
  for (int o = 16; o > 0; o >>= 1) {
    s0 += __shfl_xor(s0, o, 32);
    s1 += __shfl_xor(s1, o, 32);
    s2 += __shfl_xor(s2, o, 32);
  }
  if (lane == 0) {
    s0 += g2b[0]; s1 += g2b[1]; s2 += g2b[2];
    float m = fmaxf(s0, fmaxf(s1, s2));
    float e0 = __expf(s0 - m), e1 = __expf(s1 - m), e2 = __expf(s2 - m);
    float inv = 1.0f / (e0 + e1 + e2);
    G[row * 4 + 0] = e0 * inv;
    G[row * 4 + 1] = e1 * inv;
    G[row * 4 + 2] = e2 * inv;
    G[row * 4 + 3] = 0.0f;
  }
}

// ---------------------------------------------------------------------------
// Skinny LoRA-in: tu = V @ la_u, tf = V @ la_f  (rank 16). Thread = (b, r).
// ---------------------------------------------------------------------------
__global__ void lora_in_kernel(const bf16_t* __restrict__ V, const float* __restrict__ lau,
                               const float* __restrict__ laf, float* __restrict__ TU,
                               float* __restrict__ TF, int K) {
  int id = blockIdx.x * blockDim.x + threadIdx.x;   // B*16 threads
  int b = id >> 4, r = id & 15;
  float su = 0.f, sf = 0.f;
  const bf16_t* v = V + (long)b * K;
  for (int k = 0; k < K; ++k) {
    float vv = (float)v[k];
    su += vv * lau[k * RANK + r];
    sf += vv * laf[k * RANK + r];
  }
  TU[id] = su;
  TF[id] = sf;
}

// ---------------------------------------------------------------------------
// Head: out[b] = sigmoid(H[b,:] @ aff_w + aff_b). One wave32 per row.
// ---------------------------------------------------------------------------
__global__ void head_kernel(const bf16_t* __restrict__ H, const float* __restrict__ affw,
                            const float* __restrict__ affb, float* __restrict__ out, int K) {
  int row  = (blockIdx.x * blockDim.x + threadIdx.x) >> 5;
  int lane = threadIdx.x & 31;
  const bf16_t* h = H + (long)row * K;
  float s = 0.f;
  for (int k = lane; k < K; k += 32) s += (float)h[k] * affw[k];
#pragma unroll
  for (int o = 16; o > 0; o >>= 1) s += __shfl_xor(s, o, 32);
  if (lane == 0) out[row] = 1.0f / (1.0f + __expf(-(s + affb[0])));
}

// ---------------------------------------------------------------------------
extern "C" void kernel_launch(void* const* d_in, const int* in_sizes, int n_in,
                              void* d_out, int out_size, void* d_ws, size_t ws_size,
                              hipStream_t stream) {
  (void)in_sizes; (void)n_in; (void)out_size; (void)ws_size;

  const int*   uf0_idx = (const int*)d_in[0];
  const float* emb_uf0 = (const float*)d_in[1];
  const int*   uf1_idx = (const int*)d_in[2];
  const float* emb_uf1 = (const float*)d_in[3];
  const int*   vf0_idx = (const int*)d_in[4];
  const float* emb_vf0 = (const float*)d_in[5];
  const int*   vf1_idx = (const int*)d_in[6];
  const float* emb_vf1 = (const float*)d_in[7];

  const float *fc_w[3], *fc_b[3], *g1_w[3], *g1_b[3], *g2_w[3], *g2_b[3],
              *la_u[3], *lb_u[3], *la_f[3], *lb_f[3];
  for (int l = 0; l < 3; ++l) {
    int base = 8 + l * 10;
    fc_w[l] = (const float*)d_in[base + 0];
    fc_b[l] = (const float*)d_in[base + 1];
    g1_w[l] = (const float*)d_in[base + 2];
    g1_b[l] = (const float*)d_in[base + 3];
    g2_w[l] = (const float*)d_in[base + 4];
    g2_b[l] = (const float*)d_in[base + 5];
    la_u[l] = (const float*)d_in[base + 6];
    lb_u[l] = (const float*)d_in[base + 7];
    la_f[l] = (const float*)d_in[base + 8];
    lb_f[l] = (const float*)d_in[base + 9];
  }
  const float* aff_w = (const float*)d_in[38];
  const float* aff_b = (const float*)d_in[39];

  static const int LAY[4] = {256, 2048, 2048, 1024};

  // ---- workspace carve (256B aligned slices) ----
  char* ws = (char*)d_ws;
  size_t off = 0;
  auto alloc = [&](size_t bytes) -> char* {
    char* p = ws + off;
    off = (off + bytes + 255) & ~(size_t)255;
    return p;
  };
  bf16_t *wt_fc[3], *wt_g1[3], *wt_lbu[3], *wt_lbf[3];
  for (int l = 0; l < 3; ++l) {
    wt_fc[l]  = (bf16_t*)alloc((size_t)LAY[l] * LAY[l + 1] * 2);
    wt_g1[l]  = (bf16_t*)alloc((size_t)LAY[l] * GDIM * 2);
    wt_lbu[l] = (bf16_t*)alloc((size_t)RANK * LAY[l + 1] * 2);
    wt_lbf[l] = (bf16_t*)alloc((size_t)RANK * LAY[l + 1] * 2);
  }
  bf16_t* actA = (bf16_t*)alloc((size_t)B_ROWS * 2048 * 2);
  bf16_t* actB = (bf16_t*)alloc((size_t)B_ROWS * 2048 * 2);
  bf16_t* uG   = (bf16_t*)alloc((size_t)B_ROWS * GDIM * 2);
  float*  gate = (float*)alloc((size_t)B_ROWS * 4 * 4);
  float*  tu   = (float*)alloc((size_t)B_ROWS * RANK * 4);
  float*  tf   = (float*)alloc((size_t)B_ROWS * RANK * 4);

  // ---- weight conversion (transposed bf16) ----
  for (int l = 0; l < 3; ++l) {
    long n;
    n = (long)LAY[l] * LAY[l + 1];
    convt_kernel<<<(int)((n + 255) / 256), 256, 0, stream>>>(fc_w[l], wt_fc[l], LAY[l], LAY[l + 1]);
    n = (long)LAY[l] * GDIM;
    convt_kernel<<<(int)((n + 255) / 256), 256, 0, stream>>>(g1_w[l], wt_g1[l], LAY[l], GDIM);
    n = (long)RANK * LAY[l + 1];
    convt_kernel<<<(int)((n + 255) / 256), 256, 0, stream>>>(lb_u[l], wt_lbu[l], RANK, LAY[l + 1]);
    convt_kernel<<<(int)((n + 255) / 256), 256, 0, stream>>>(lb_f[l], wt_lbf[l], RANK, LAY[l + 1]);
  }

  // ---- embedding gather -> v0 [B,256] bf16 ----
  embed_gather_kernel<<<(B_ROWS * 256) / 256, 256, 0, stream>>>(
      uf0_idx, uf1_idx, vf0_idx, vf1_idx, emb_uf0, emb_uf1, emb_vf0, emb_vf1, actA);

  // ---- 3 gated-LoRA layers ----
  bf16_t* vin = actA;
  bf16_t* vout = actB;
  for (int l = 0; l < 3; ++l) {
    const int K = LAY[l], N = LAY[l + 1];

    dim3 gGate(GDIM / BN, B_ROWS / BM);
    gemm_bf16_kernel<0><<<gGate, 256, 0, stream>>>(
        vin, wt_g1[l], g1_b[l], uG, K, GDIM, nullptr, nullptr, nullptr, nullptr, nullptr);

    gate_kernel<<<B_ROWS / 8, 256, 0, stream>>>(uG, g2_w[l], g2_b[l], gate);

    lora_in_kernel<<<(B_ROWS * RANK) / 256, 256, 0, stream>>>(vin, la_u[l], la_f[l], tu, tf, K);

    dim3 gFc(N / BN, B_ROWS / BM);
    gemm_bf16_kernel<1><<<gFc, 256, 0, stream>>>(
        vin, wt_fc[l], fc_b[l], vout, K, N, gate, tu, tf, wt_lbu[l], wt_lbf[l]);

    bf16_t* t = vin; vin = vout; vout = t;
  }

  // ---- head: sigmoid(h @ aff_w + aff_b) ----
  head_kernel<<<B_ROWS / 8, 256, 0, stream>>>(vin, aff_w, aff_b, (float*)d_out, LAY[3]);
}
